// SchematicLSTM_2190433321446
// MI455X (gfx1250) — matrix operand order repaired and checked by reading, not compiled, not run on hardware
//
#include <hip/hip_runtime.h>
#include <math.h>

#define BATCH  32
#define SEQ    1024
#define VOCAB  2048
#define HIDDEN 1024

typedef __attribute__((ext_vector_type(16))) __bf16 v16bf;
typedef __attribute__((ext_vector_type(8)))  float  v8f;

union Frag32B { uint4 q[2]; v16bf v; };

// --- WMMA fragment loaders (wave32, ISA 7.12.2 layouts) -------------------
// A-matrix 16x32 bf16: lane = (m = lane%16, half = lane/16)
//   frag elems 0..7  hold K = kb + half*8      .. +7   (contiguous)
//   frag elems 8..15 hold K = kb + 16 + half*8 .. +7   (contiguous)
static __device__ __forceinline__ v16bf load_a_frag(const __bf16* row, int kb, int half) {
  Frag32B f;
  f.q[0] = *(const uint4*)(row + kb + half * 8);
  f.q[1] = *(const uint4*)(row + kb + 16 + half * 8);
  return f.v;
}
// B-matrix 32x16 bf16: lane = (n = lane%16, half)
//   frag elems 0..7 hold K = kb + half*8 .. +7, elems 8..15 hold K = kb + 16 + half*8 .. +7
//   (contiguous in the W row, since B[k][n] = W[n][k] and W rows are contiguous in k)
static __device__ __forceinline__ v16bf load_b_frag(const __bf16* row, int kb, int half) {
  Frag32B f;
  f.q[0] = *(const uint4*)(row + kb + half * 8);
  f.q[1] = *(const uint4*)(row + kb + 16 + half * 8);
  return f.v;
}

static __device__ __forceinline__ v8f wmma_bf16(v16bf a, v16bf b, v8f c) {
  return __builtin_amdgcn_wmma_f32_16x16x32_bf16(false, a, false, b,
                                                 (short)0, c, false, false);
}

// --- prep kernels ----------------------------------------------------------

// E[v][h] = bf16(W_ih[h][v])  (transpose+convert via LDS tile)
__global__ void __launch_bounds__(256)
k_transpose_wih(const float* __restrict__ W, __bf16* __restrict__ E) {
  __shared__ float tile[16][17];
  const int tx = threadIdx.x, ty = threadIdx.y;
  const int v0 = blockIdx.x * 16, h0 = blockIdx.y * 16;
  tile[ty][tx] = W[(size_t)(h0 + ty) * VOCAB + (v0 + tx)]; // coalesced over v
  __syncthreads();
  E[(size_t)(v0 + ty) * HIDDEN + (h0 + tx)] = (__bf16)tile[tx][ty]; // coalesced over h
}

__global__ void __launch_bounds__(256)
k_f32_to_bf16(const float* __restrict__ src, __bf16* __restrict__ dst, int n) {
  int i = blockIdx.x * blockDim.x + threadIdx.x;
  if (i < n) dst[i] = (__bf16)src[i];
}

__global__ void __launch_bounds__(256)
k_bias_sum(const float* __restrict__ a, const float* __restrict__ b,
           float* __restrict__ c, int n) {
  int i = blockIdx.x * blockDim.x + threadIdx.x;
  if (i < n) c[i] = a[i] + b[i];
}

// --- recurrent step: Hout = tanh(E[tok] + bias + Hprev @ Whh^T) ------------
// one wave per block; block computes the full batch (M=32) x 16 N-columns
__global__ void __launch_bounds__(32)
k_rnn_step(const __bf16* __restrict__ Hprev, const __bf16* __restrict__ E,
           const __bf16* __restrict__ Whh, const float* __restrict__ bias,
           const int* __restrict__ tokens, int t,
           __bf16* __restrict__ Hout, float* __restrict__ HoutF) {
  const int lane = threadIdx.x;
  const int nl = lane & 15, half = lane >> 4;
  const int j0 = blockIdx.x * 16;

  __shared__ int tok[BATCH];
  tok[lane] = tokens[lane * SEQ + t]; // inputs[b][t]
  __syncthreads();

  const __bf16* arow0 = Hprev + (size_t)nl * HIDDEN;        // batch rows 0..15
  const __bf16* arow1 = Hprev + (size_t)(nl + 16) * HIDDEN; // batch rows 16..31
  const __bf16* brow  = Whh + (size_t)(j0 + nl) * HIDDEN;   // Whh row j = B col

  v8f acc0 = {}, acc1 = {};
  for (int kb = 0; kb < HIDDEN; kb += 32) {
    v16bf b  = load_b_frag(brow, kb, half);
    v16bf a0 = load_a_frag(arow0, kb, half);
    v16bf a1 = load_a_frag(arow1, kb, half);
    acc0 = wmma_bf16(a0, b, acc0);
    acc1 = wmma_bf16(a1, b, acc1);
  }

  // C/D layout: lane nl gives N; VGPR r gives M = half*8 + r (acc0), +16 (acc1)
  const int n = j0 + nl;
  const float bn = bias[n];
#pragma unroll
  for (int r = 0; r < 8; ++r) {
    const int m0 = half * 8 + r;
    const int m1 = m0 + 16;
    const float x0 = (float)E[(size_t)tok[m0] * HIDDEN + n] + bn;
    const float x1 = (float)E[(size_t)tok[m1] * HIDDEN + n] + bn;
    const float h0 = tanhf(acc0[r] + x0);
    const float h1 = tanhf(acc1[r] + x1);
    Hout[(size_t)m0 * HIDDEN + n] = (__bf16)h0;
    Hout[(size_t)m1 * HIDDEN + n] = (__bf16)h1;
    if (HoutF) {                      // only the final step emits fp32 hT
      HoutF[(size_t)m0 * HIDDEN + n] = h0;
      HoutF[(size_t)m1 * HIDDEN + n] = h1;
    }
  }
}

// --- output projection: out = Y @ W_lin^T + b_lin --------------------------
// block = 4 waves; each wave computes a 64M x 32N tile (4 A-frags, 2 B-frags,
// 8 WMMAs per k-step -> 1.5 loads/WMMA, halving L2 fetch per FLOP)
__global__ void __launch_bounds__(128)
k_out_gemm(const __bf16* __restrict__ Y, const __bf16* __restrict__ Wl,
           const float* __restrict__ blin, float* __restrict__ out) {
  const int lane = threadIdx.x & 31;
  const int wv   = threadIdx.x >> 5;
  const int nl = lane & 15, half = lane >> 4;
  const int n0 = blockIdx.x * 128 + wv * 32;   // wave-private 32-wide N tile
  const size_t m0 = (size_t)blockIdx.y * 64;   // block 64-row M tile

  const __bf16* arow[4];
#pragma unroll
  for (int i = 0; i < 4; ++i) arow[i] = Y + (m0 + i * 16 + nl) * HIDDEN;
  const __bf16* brow[2];
#pragma unroll
  for (int j = 0; j < 2; ++j) brow[j] = Wl + (size_t)(n0 + j * 16 + nl) * HIDDEN;

  v8f acc[4][2] = {};
  for (int kb = 0; kb < HIDDEN; kb += 32) {
    __builtin_prefetch(brow[0] + kb + 256, 0, 0); // global_prefetch_b8
    v16bf b0 = load_b_frag(brow[0], kb, half);
    v16bf b1 = load_b_frag(brow[1], kb, half);
#pragma unroll
    for (int i = 0; i < 4; ++i) {
      v16bf a = load_a_frag(arow[i], kb, half);
      acc[i][0] = wmma_bf16(a, b0, acc[i][0]);
      acc[i][1] = wmma_bf16(a, b1, acc[i][1]);
    }
  }

#pragma unroll
  for (int j = 0; j < 2; ++j) {
    const int n = n0 + j * 16 + nl;
    const float bn = blin[n];
#pragma unroll
    for (int i = 0; i < 4; ++i) {
#pragma unroll
      for (int r = 0; r < 8; ++r) {
        const size_t m = m0 + i * 16 + half * 8 + r;
        out[m * VOCAB + n] = acc[i][j][r] + bn;
      }
    }
  }
}

// --- host launcher ---------------------------------------------------------
extern "C" void kernel_launch(void* const* d_in, const int* in_sizes, int n_in,
                              void* d_out, int out_size, void* d_ws, size_t ws_size,
                              hipStream_t stream) {
  (void)in_sizes; (void)n_in; (void)out_size; (void)ws_size;

  const int*   tokens = (const int*)d_in[0];   // (BATCH, SEQ)
  const float* state  = (const float*)d_in[1]; // (1, BATCH, HIDDEN)
  const float* W_ih   = (const float*)d_in[2]; // (HIDDEN, VOCAB)
  const float* b_ih   = (const float*)d_in[3];
  const float* W_hh   = (const float*)d_in[4]; // (HIDDEN, HIDDEN)
  const float* b_hh   = (const float*)d_in[5];
  const float* W_lin  = (const float*)d_in[6]; // (VOCAB, HIDDEN)
  const float* b_lin  = (const float*)d_in[7];

  float* out = (float*)d_out;                          // (SEQ*BATCH, VOCAB)
  float* hT  = out + (size_t)SEQ * BATCH * VOCAB;      // (1, BATCH, HIDDEN)

  // workspace carve-up (all sizes 256B-aligned by construction)
  char* p = (char*)d_ws;
  __bf16* E    = (__bf16*)p; p += (size_t)VOCAB * HIDDEN * sizeof(__bf16);  // 4MB
  __bf16* Whh  = (__bf16*)p; p += (size_t)HIDDEN * HIDDEN * sizeof(__bf16); // 2MB
  __bf16* Wl   = (__bf16*)p; p += (size_t)VOCAB * HIDDEN * sizeof(__bf16);  // 4MB
  float*  bias = (float*)p;  p += (size_t)HIDDEN * sizeof(float);           // 4KB
  __bf16* H0   = (__bf16*)p; p += (size_t)BATCH * HIDDEN * sizeof(__bf16);  // 64KB
  __bf16* Ybf  = (__bf16*)p;                                                // 64MB

  // phase 0: precision/layout prep
  k_transpose_wih<<<dim3(VOCAB / 16, HIDDEN / 16), dim3(16, 16), 0, stream>>>(W_ih, E);
  k_f32_to_bf16<<<(HIDDEN * HIDDEN) / 256, 256, 0, stream>>>(W_hh, Whh, HIDDEN * HIDDEN);
  k_f32_to_bf16<<<(VOCAB * HIDDEN) / 256, 256, 0, stream>>>(W_lin, Wl, VOCAB * HIDDEN);
  k_f32_to_bf16<<<(BATCH * HIDDEN) / 256, 256, 0, stream>>>(state, H0, BATCH * HIDDEN);
  k_bias_sum<<<HIDDEN / 256, 256, 0, stream>>>(b_ih, b_hh, bias, HIDDEN);

  // phase 1: sequential recurrence; Y[t] doubles as h_{t} storage
  for (int t = 0; t < SEQ; ++t) {
    const __bf16* Hprev = (t == 0) ? H0 : (Ybf + (size_t)(t - 1) * BATCH * HIDDEN);
    __bf16* Hout = Ybf + (size_t)t * BATCH * HIDDEN;
    float* hF = (t == SEQ - 1) ? hT : nullptr;
    k_rnn_step<<<HIDDEN / 16, 32, 0, stream>>>(Hprev, E, Whh, bias, tokens, t, Hout, hF);
  }

  // phase 2: big output GEMM (grid: N tiles of 128, M tiles of 64)
  k_out_gemm<<<dim3(VOCAB / 128, (SEQ * BATCH) / 64), 128, 0, stream>>>(Ybf, Wl, b_lin, out);
}
